// MolGAT_77171972375184
// MI455X (gfx1250) — matrix operand order repaired and checked by reading, not compiled, not run on hardware
//
#include <hip/hip_runtime.h>
#include <hip/hip_bf16.h>
#include <float.h>

#define N_NODES 20000
#define N_EDGES 320000
#define N_GRAPH 128
#define F_NODE  64
#define F_EDGE  16
#define HID     128
#define HEADS   2
#define HC      (HEADS*HID)   /* 256 */
#define NEG_SLOPE 0.2f

typedef __attribute__((ext_vector_type(16))) __bf16 v16bf;
typedef __attribute__((ext_vector_type(8)))  float  v8f;

// ---------- monotone float <-> uint encoding for atomic max ----------
__device__ __forceinline__ unsigned f2ord(float f){
  unsigned u = __float_as_uint(f);
  return (u & 0x80000000u) ? ~u : (u | 0x80000000u);
}
__device__ __forceinline__ float ord2f(unsigned u){
  return (u & 0x80000000u) ? __uint_as_float(u & 0x7FFFFFFFu) : __uint_as_float(~u);
}
#define ENC_NEGMAX 0x00800000u   /* f2ord(-FLT_MAX) */

__global__ void fill_u32(unsigned* __restrict__ p, unsigned v, int n){
  int i = blockIdx.x * blockDim.x + threadIdx.x;
  if (i < n) p[i] = v;
}

// ---------- pack weight B[K,Nc] (fp32, row-major) into WMMA-fragment order ----
// Layout: [kblock][ntile][lane][16 bf16]  (one 32-byte chunk per lane per fragment)
// Fragment layout per ISA: lanes 0-15 hold K=k0..k0+7 in v0..3, K=k0+16..k0+23 in v4..7; lanes 16-31 +8.
__global__ void pack_B_bf16(const float* __restrict__ B, __bf16* __restrict__ Bp,
                            int K, int Nc){
  int t = blockIdx.x * blockDim.x + threadIdx.x;
  int total = (K >> 5) * (Nc >> 4) * 32;
  if (t >= total) return;
  int lane = t & 31;
  int tile = (t >> 5) % (Nc >> 4);
  int kblk = (t >> 5) / (Nc >> 4);
  int sel = lane >> 4, nn = lane & 15;
  int n = tile * 16 + nn;
  __bf16* out = Bp + (size_t)t * 16;
#pragma unroll
  for (int r = 0; r < 8; ++r) {
    int kb = kblk * 32 + (r >> 2) * 16 + sel * 8 + 2 * (r & 3);
    out[2*r]   = (__bf16)B[(size_t)kb * Nc + n];
    out[2*r+1] = (__bf16)B[(size_t)(kb + 1) * Nc + n];
  }
}

// ---------- generic GEMM: C[M,Nc] = act(A[M,K] @ B[K,Nc] + bias) ----------
// fp32 A (converted on the fly), pre-packed bf16 B fragments, fp32 accumulate.
// One 16(M) x 64(N) tile per wave: 4 accumulators share one A fragment.
// Requirements (all call sites satisfy): M%16==0, Nc%64==0, K%32==0.
template<int K>
__global__ void gemm_bf16_wmma(const float* __restrict__ A, const __bf16* __restrict__ Bp,
                               const float* __restrict__ bias, float* __restrict__ C,
                               int M, int Nc, int relu)
{
  const int wave = threadIdx.x >> 5;
  const int lane = threadIdx.x & 31;
  const int lh   = lane & 15;
  const int sel  = lane >> 4;
  const int groups = Nc >> 6;   // 64-wide column groups
  const int tilesN = Nc >> 4;
  const int tilesM = M >> 4;
  const int t = blockIdx.x * (blockDim.x >> 5) + wave;
  const int tile_m = t / groups;
  const int grp    = t - tile_m * groups;
  if (tile_m >= tilesM) return;      // wave-uniform: EXEC all-ones around WMMA

  const int mA = tile_m * 16 + lh;
  const int n0 = grp * 64 + lh;
  const float* __restrict__ Arow = A + (size_t)mA * K;

  v8f acc0 = {}, acc1 = {}, acc2 = {}, acc3 = {};

#pragma unroll
  for (int k0 = 0; k0 < K; k0 += 32) {
    // ---- A fragment: two contiguous 8-float spans -> 4x global_load_b128 ----
    // layout: lanes 0-15: v0..3 = K 0..7, v4..7 = K 16..23; lanes 16-31: +8
    float4 a0 = *(const float4*)(Arow + k0 + sel * 8);
    float4 a1 = *(const float4*)(Arow + k0 + sel * 8 + 4);
    float4 a2 = *(const float4*)(Arow + k0 + 16 + sel * 8);
    float4 a3 = *(const float4*)(Arow + k0 + 16 + sel * 8 + 4);
    v16bf a;
    a[0]  = (__bf16)a0.x; a[1]  = (__bf16)a0.y; a[2]  = (__bf16)a0.z; a[3]  = (__bf16)a0.w;
    a[4]  = (__bf16)a1.x; a[5]  = (__bf16)a1.y; a[6]  = (__bf16)a1.z; a[7]  = (__bf16)a1.w;
    a[8]  = (__bf16)a2.x; a[9]  = (__bf16)a2.y; a[10] = (__bf16)a2.z; a[11] = (__bf16)a2.w;
    a[12] = (__bf16)a3.x; a[13] = (__bf16)a3.y; a[14] = (__bf16)a3.z; a[15] = (__bf16)a3.w;

    // ---- B fragments: one aligned 32-byte vector load each, no conversion ----
    const __bf16* bp = Bp + (((size_t)(k0 >> 5) * tilesN + (size_t)grp * 4) * 32 + lane) * 16;
    v16bf b0 = *(const v16bf*)(bp);
    v16bf b1 = *(const v16bf*)(bp + 512);    // next ntile: 32 lanes * 16 elems
    v16bf b2 = *(const v16bf*)(bp + 1024);
    v16bf b3 = *(const v16bf*)(bp + 1536);

    acc0 = __builtin_amdgcn_wmma_f32_16x16x32_bf16(false, a, false, b0, (short)0, acc0, false, false);
    acc1 = __builtin_amdgcn_wmma_f32_16x16x32_bf16(false, a, false, b1, (short)0, acc1, false, false);
    acc2 = __builtin_amdgcn_wmma_f32_16x16x32_bf16(false, a, false, b2, (short)0, acc2, false, false);
    acc3 = __builtin_amdgcn_wmma_f32_16x16x32_bf16(false, a, false, b3, (short)0, acc3, false, false);
  }

  // ---- store: C/D layout: vgpr r -> row r (lanes 0-15), row r+8 (lanes 16-31) ----
  const int mBase = tile_m * 16 + sel * 8;
#pragma unroll
  for (int c = 0; c < 4; ++c) {
    const v8f acc = (c == 0) ? acc0 : (c == 1) ? acc1 : (c == 2) ? acc2 : acc3;
    const int n = n0 + 16 * c;
    const float bv = bias ? bias[n] : 0.f;
#pragma unroll
    for (int r = 0; r < 8; ++r) {
      float v = acc[r] + bv;
      if (relu) v = fmaxf(v, 0.f);
      C[(size_t)(mBase + r) * Nc + n] = v;
    }
  }
}

// ---------- per-node attention dots: di = <h, ai>, dj = <h, aj> ----------
__global__ void node_dots(const float* __restrict__ H, const float* __restrict__ ai,
                          const float* __restrict__ aj, float* __restrict__ di,
                          float* __restrict__ dj, int n){
  int t = blockIdx.x * blockDim.x + threadIdx.x;
  if (t >= n * HEADS) return;
  int node = t >> 1, hd = t & 1;
  const float* hp  = H  + (size_t)node * HC + hd * HID;
  const float* aip = ai + hd * HID;
  const float* ajp = aj + hd * HID;
  float si = 0.f, sj = 0.f;
  for (int c = 0; c < HID; ++c) { float hv = hp[c]; si += hv * aip[c]; sj += hv * ajp[c]; }
  di[t] = si; dj[t] = sj;
}

// ---------- wej[h] = We @ aj[h]  (folds edge features into 16-vector) ----------
__global__ void compute_wej(const float* __restrict__ We, const float* __restrict__ aj,
                            float* __restrict__ wej){
  int t = threadIdx.x;
  if (t >= HEADS * F_EDGE) return;
  int hd = t >> 4, k = t & 15;
  float s = 0.f;
  for (int c = 0; c < HID; ++c) s += We[(size_t)k * HC + hd * HID + c] * aj[hd * HID + c];
  wej[t] = s;
}

// ---------- edge logits + segment max ----------
__global__ void edge_logits(const int* __restrict__ src, const int* __restrict__ dst,
                            const float* __restrict__ ea, const float* __restrict__ di,
                            const float* __restrict__ dj, const float* __restrict__ wej,
                            float* __restrict__ lg, unsigned* __restrict__ lmax, int E){
  int e = blockIdx.x * blockDim.x + threadIdx.x;
  if (e >= E) return;
  int s = src[e], d = dst[e];
  float eav[F_EDGE];
#pragma unroll
  for (int k = 0; k < F_EDGE; ++k) eav[k] = ea[(size_t)e * F_EDGE + k];
#pragma unroll
  for (int hd = 0; hd < HEADS; ++hd) {
    float v = di[d * HEADS + hd] + dj[s * HEADS + hd];
#pragma unroll
    for (int k = 0; k < F_EDGE; ++k) v += eav[k] * wej[hd * F_EDGE + k];
    v = (v > 0.f) ? v : NEG_SLOPE * v;
    lg[(size_t)e * HEADS + hd] = v;
    atomicMax(&lmax[d * HEADS + hd], f2ord(v));
  }
}

// ---------- p = exp(logit - max), segment-sum denominators ----------
__global__ void edge_p(const int* __restrict__ dst, float* __restrict__ lg,
                       const unsigned* __restrict__ lmax, float* __restrict__ den, int E){
  int t = blockIdx.x * blockDim.x + threadIdx.x;
  if (t >= E * HEADS) return;
  int e = t >> 1, hd = t & 1;
  int d = dst[e];
  float p = __expf(lg[t] - ord2f(lmax[d * HEADS + hd]));
  lg[t] = p;
  atomicAdd(&den[d * HEADS + hd], p);
}

// ---------- weighted scatter-add; recompute e = edge_attr@We from LDS-staged We ----------
__global__ void edge_aggregate(const int* __restrict__ src, const int* __restrict__ dst,
                               const float* __restrict__ ea, const float* __restrict__ We,
                               const float* __restrict__ H, const float* __restrict__ p,
                               const float* __restrict__ den, float* __restrict__ outacc,
                               int E){
  __shared__ float sWe[F_EDGE * HC];             // 16 KB of the WGP's 320 KB LDS
  for (int i = threadIdx.x; i < F_EDGE * HC; i += blockDim.x) sWe[i] = We[i];
  __syncthreads();
  int e = blockIdx.x * (blockDim.x >> 5) + (threadIdx.x >> 5);
  if (e >= E) return;
  int lane = threadIdx.x & 31;
  int s = src[e], d = dst[e];
  float eav[F_EDGE];
#pragma unroll
  for (int k = 0; k < F_EDGE; ++k) eav[k] = ea[(size_t)e * F_EDGE + k];
  float a0 = p[(size_t)e * HEADS]     / (den[d * HEADS]     + 1e-16f);
  float a1 = p[(size_t)e * HEADS + 1] / (den[d * HEADS + 1] + 1e-16f);
#pragma unroll
  for (int j = 0; j < HC / 32; ++j) {
    int ch = lane + 32 * j;                       // coalesced across the wave
    float ef = 0.f;
#pragma unroll
    for (int k = 0; k < F_EDGE; ++k) ef += eav[k] * sWe[k * HC + ch];
    float mv = H[(size_t)s * HC + ch] + ef;
    float al = (ch < HID) ? a0 : a1;
    atomicAdd(&outacc[(size_t)d * HC + ch], al * mv);
  }
}

// ---------- head-mean + bias + ReLU -> next layer's node features ----------
__global__ void finalize_node(const float* __restrict__ outacc, const float* __restrict__ b,
                              float* __restrict__ xn, int n){
  int t = blockIdx.x * blockDim.x + threadIdx.x;
  if (t >= n * HID) return;
  int node = t >> 7, c = t & (HID - 1);
  float v = 0.5f * (outacc[(size_t)node * HC + c] + outacc[(size_t)node * HC + HID + c]) + b[c];
  xn[t] = fmaxf(v, 0.f);
}

// ---------- per-graph max/mean pooling ----------
__global__ void pool_scatter(const float* __restrict__ x, const int* __restrict__ batch,
                             unsigned* __restrict__ gmax, float* __restrict__ gsum,
                             float* __restrict__ cnt, int n){
  int t = blockIdx.x * blockDim.x + threadIdx.x;
  if (t >= n * HID) return;
  int node = t >> 7, c = t & (HID - 1);
  int g = batch[node];
  float v = x[t];
  atomicMax(&gmax[g * HID + c], f2ord(v));
  atomicAdd(&gsum[g * HID + c], v);
  if (c == 0) atomicAdd(&cnt[g], 1.f);
}

__global__ void pool_finalize(const unsigned* __restrict__ gmax, const float* __restrict__ gsum,
                              const float* __restrict__ cnt, float* __restrict__ hcat){
  int t = blockIdx.x * blockDim.x + threadIdx.x;
  if (t >= N_GRAPH * HID) return;
  int g = t >> 7, c = t & (HID - 1);
  hcat[g * 2 * HID + c]       = ord2f(gmax[t]);
  hcat[g * 2 * HID + HID + c] = gsum[t] / fmaxf(cnt[g], 1.f);
}

// ---------- 128x128x1 output head ----------
__global__ void final_out(const float* __restrict__ fc2, const float* __restrict__ W,
                          const float* __restrict__ b, float* __restrict__ out){
  int g = blockIdx.x * blockDim.x + threadIdx.x;
  if (g >= N_GRAPH) return;
  float s = b[0];
  for (int c = 0; c < HID; ++c) s += fc2[g * HID + c] * W[c];
  out[g] = s;
}

extern "C" void kernel_launch(void* const* d_in, const int* in_sizes, int n_in,
                              void* d_out, int out_size, void* d_ws, size_t ws_size,
                              hipStream_t stream) {
  const float* x      = (const float*)d_in[0];
  const int*   eidx   = (const int*)d_in[1];
  const int*   batch  = (const int*)d_in[2];
  const float* ea     = (const float*)d_in[3];
  const float* convW[3]  = {(const float*)d_in[4],  (const float*)d_in[5],  (const float*)d_in[6]};
  const float* convWe[3] = {(const float*)d_in[7],  (const float*)d_in[8],  (const float*)d_in[9]};
  const float* convAi[3] = {(const float*)d_in[10], (const float*)d_in[11], (const float*)d_in[12]};
  const float* convAj[3] = {(const float*)d_in[13], (const float*)d_in[14], (const float*)d_in[15]};
  const float* convB[3]  = {(const float*)d_in[16], (const float*)d_in[17], (const float*)d_in[18]};
  const float* fcW[2]    = {(const float*)d_in[19], (const float*)d_in[20]};
  const float* fcB[2]    = {(const float*)d_in[21], (const float*)d_in[22]};
  const float* outW      = (const float*)d_in[23];
  const float* outB      = (const float*)d_in[24];

  const int* src = eidx;
  const int* dst = eidx + N_EDGES;

  // ---- workspace carve-out (~55 MB) ----
  float* ws = (float*)d_ws;
  float* H       = ws; ws += (size_t)N_NODES * HC;
  float* xbuf    = ws; ws += (size_t)N_NODES * HID;
  float* di      = ws; ws += N_NODES * HEADS;
  float* dj      = ws; ws += N_NODES * HEADS;
  float* pbuf    = ws; ws += (size_t)N_EDGES * HEADS;
  unsigned* lmax = (unsigned*)ws; ws += N_NODES * HEADS;
  float* den     = ws; ws += N_NODES * HEADS;
  float* outacc  = ws; ws += (size_t)N_NODES * HC;
  float* wej     = ws; ws += HEADS * F_EDGE;
  unsigned* gmaxb = (unsigned*)ws; ws += N_GRAPH * HID;
  float* gsum    = ws; ws += N_GRAPH * HID;
  float* cnt     = ws; ws += N_GRAPH;
  float* hcat    = ws; ws += N_GRAPH * 2 * HID;
  float* fc1     = ws; ws += N_GRAPH * 2 * HID;
  float* fc2     = ws; ws += N_GRAPH * HID;
  __bf16* Bp     = (__bf16*)ws; ws += (2 * HID * HC) / 2;   // 256x256 bf16 = 32768 floats

  const int B = 256;
  auto nb = [](long n, int b){ return (int)((n + b - 1) / b); };

  for (int l = 0; l < 3; ++l) {
    const float* xin = (l == 0) ? x : xbuf;
    const int K = (l == 0) ? F_NODE : HID;

    // pack weights into WMMA fragment order (bf16), then H = xin @ W
    pack_B_bf16<<<nb((K / 32) * (HC / 16) * 32, B), B, 0, stream>>>(convW[l], Bp, K, HC);
    int tiles = (N_NODES / 16) * (HC / 64);
    if (l == 0)
      gemm_bf16_wmma<F_NODE><<<nb(tiles, 8), 256, 0, stream>>>(xin, Bp, nullptr, H,
                                                               N_NODES, HC, 0);
    else
      gemm_bf16_wmma<HID><<<nb(tiles, 8), 256, 0, stream>>>(xin, Bp, nullptr, H,
                                                            N_NODES, HC, 0);

    node_dots<<<nb(N_NODES * HEADS, B), B, 0, stream>>>(H, convAi[l], convAj[l], di, dj, N_NODES);
    compute_wej<<<1, 32, 0, stream>>>(convWe[l], convAj[l], wej);

    fill_u32<<<nb(N_NODES * HEADS, B), B, 0, stream>>>(lmax, ENC_NEGMAX, N_NODES * HEADS);
    fill_u32<<<nb(N_NODES * HEADS, B), B, 0, stream>>>((unsigned*)den, 0u, N_NODES * HEADS);
    fill_u32<<<nb((long)N_NODES * HC, B), B, 0, stream>>>((unsigned*)outacc, 0u, N_NODES * HC);

    edge_logits<<<nb(N_EDGES, B), B, 0, stream>>>(src, dst, ea, di, dj, wej, pbuf, lmax, N_EDGES);
    edge_p<<<nb(N_EDGES * HEADS, B), B, 0, stream>>>(dst, pbuf, lmax, den, N_EDGES);
    edge_aggregate<<<nb(N_EDGES, 8), 256, 0, stream>>>(src, dst, ea, convWe[l], H, pbuf, den,
                                                       outacc, N_EDGES);
    finalize_node<<<nb((long)N_NODES * HID, B), B, 0, stream>>>(outacc, convB[l], xbuf, N_NODES);
  }

  fill_u32<<<nb(N_GRAPH * HID, B), B, 0, stream>>>(gmaxb, ENC_NEGMAX, N_GRAPH * HID);
  fill_u32<<<nb(N_GRAPH * HID, B), B, 0, stream>>>((unsigned*)gsum, 0u, N_GRAPH * HID);
  fill_u32<<<nb(N_GRAPH, B), B, 0, stream>>>((unsigned*)cnt, 0u, N_GRAPH);
  pool_scatter<<<nb((long)N_NODES * HID, B), B, 0, stream>>>(xbuf, batch, gmaxb, gsum, cnt, N_NODES);
  pool_finalize<<<nb(N_GRAPH * HID, B), B, 0, stream>>>(gmaxb, gsum, cnt, hcat);

  // FC layers via WMMA (M=128, K=256)
  pack_B_bf16<<<nb((2 * HID / 32) * (2 * HID / 16) * 32, B), B, 0, stream>>>(fcW[0], Bp,
                                                                             2 * HID, 2 * HID);
  int tiles = (N_GRAPH / 16) * (2 * HID / 64);
  gemm_bf16_wmma<2 * HID><<<nb(tiles, 8), 256, 0, stream>>>(hcat, Bp, fcB[0], fc1,
                                                            N_GRAPH, 2 * HID, 1);
  pack_B_bf16<<<nb((2 * HID / 32) * (HID / 16) * 32, B), B, 0, stream>>>(fcW[1], Bp,
                                                                         2 * HID, HID);
  tiles = (N_GRAPH / 16) * (HID / 64);
  gemm_bf16_wmma<2 * HID><<<nb(tiles, 8), 256, 0, stream>>>(fc1, Bp, fcB[1], fc2,
                                                            N_GRAPH, HID, 1);
  final_out<<<nb(N_GRAPH, 128), 128, 0, stream>>>(fc2, outW, outB, (float*)d_out);
}
